// MaskedCausalAttention_53850299957339
// MI455X (gfx1250) — matrix operand order repaired and checked by reading, not compiled, not run on hardware
//
#include <hip/hip_runtime.h>

// ---------------------------------------------------------------------------
// Types matching the gfx1250 WMMA / TDM builtin signatures
// ---------------------------------------------------------------------------
typedef __bf16 bf16;
typedef __attribute__((ext_vector_type(16))) __bf16 v16bf;
typedef __attribute__((ext_vector_type(8)))  __bf16 v8bf;
typedef __attribute__((ext_vector_type(8)))  float  v8f;
typedef __attribute__((ext_vector_type(4)))  unsigned int u32x4;
typedef __attribute__((ext_vector_type(8)))  int  i32x8;
typedef __attribute__((ext_vector_type(4)))  int  i32x4;

#define TT     2048
#define CC     1024
#define NHEAD  16
#define HDIM   64
#define MROWS  4096   // B*T

#if __has_builtin(__builtin_amdgcn_tensor_load_to_lds)
#define HAVE_TDM 1
#else
#define HAVE_TDM 0
#endif

__device__ __forceinline__ v16bf cat8(v8bf lo, v8bf hi) {
  return __builtin_shufflevector(lo, hi, 0,1,2,3,4,5,6,7,8,9,10,11,12,13,14,15);
}
__device__ __forceinline__ v8f wmma_bf16(v16bf a, v16bf b, v8f c) {
  // (neg_a, A, neg_b, B, c_mod, C, reuse_a, reuse_b)
  return __builtin_amdgcn_wmma_f32_16x16x32_bf16(false, a, false, b, (short)0, c, false, false);
}
__device__ __forceinline__ v8f v8f_zero() {
  v8f z;
#pragma unroll
  for (int i = 0; i < 8; ++i) z[i] = 0.0f;
  return z;
}

// ---------------------------------------------------------------------------
// fp32 -> bf16 conversion (8 elements / thread)
// ---------------------------------------------------------------------------
__global__ void __launch_bounds__(256) f32_to_bf16_kernel(const float* __restrict__ in,
                                                          bf16* __restrict__ out, int n) {
  int i = (blockIdx.x * 256 + threadIdx.x) * 8;
  if (i + 8 <= n) {
    float4 a = *(const float4*)(in + i);
    float4 b = *(const float4*)(in + i + 4);
    v8bf o;
    o[0] = (bf16)a.x; o[1] = (bf16)a.y; o[2] = (bf16)a.z; o[3] = (bf16)a.w;
    o[4] = (bf16)b.x; o[5] = (bf16)b.y; o[6] = (bf16)b.z; o[7] = (bf16)b.w;
    *(v8bf*)(out + i) = o;
  }
}

// ---------------------------------------------------------------------------
// bf16 GEMM: C[M,N] = (A[M,K] @ W[K,N] + bias[N]) * scale
// block tile 128x128, BK=32, 256 threads = 8 waves (4x2 grid),
// each wave computes 32x64 = 2x4 WMMA tiles -> 8 WMMAs per 6 fragment loads.
// ---------------------------------------------------------------------------
#define BM 128
#define BN 128
#define BK 32
#define LDT 40   // BK + 8 pad (bf16 elems), keeps rows 16B-aligned (80B stride)

template <int OUT_F32>
__global__ void __launch_bounds__(256)
gemm_bf16_kernel(const bf16* __restrict__ A, const bf16* __restrict__ W,
                 const float* __restrict__ bias, void* __restrict__ Cout,
                 int Mtot, int Ntot, int Ktot, float scale) {
  __shared__ __align__(16) bf16 As[BM][LDT];
  __shared__ __align__(16) bf16 Bs[BN][LDT];   // stored transposed: [n][k]

  const int tid  = threadIdx.x;
  const int lane = tid & 31;
  const int wave = tid >> 5;
  const int m0 = blockIdx.y * BM;
  const int n0 = blockIdx.x * BN;

  const int wr = (wave & 3) * 32;   // wave row offset in block (4 row-groups)
  const int wc = (wave >> 2) * 64;  // wave col offset in block (2 col-groups)

  // global->LDS loader indices
  const int ar  = tid >> 1;         // 0..127 row of A tile
  const int ak  = (tid & 1) * 16;   // 0 or 16
  const int bkr = tid >> 3;         // 0..31 k-row of W tile
  const int bnc = (tid & 7) * 16;   // 0..112 col

  v8f acc[2][4];
#pragma unroll
  for (int i = 0; i < 2; ++i)
#pragma unroll
    for (int j = 0; j < 4; ++j) acc[i][j] = v8f_zero();

  const int lr    = lane & 15;
  const int khalf = (lane >> 4) * 8;
  const int ks    = (lane >> 4) * 16;

  for (int kt = 0; kt < Ktot; kt += BK) {
    // --- stage tiles into LDS ---
    const bf16* ga = A + (size_t)(m0 + ar) * Ktot + kt + ak;
    *(v8bf*)&As[ar][ak]     = *(const v8bf*)(ga);
    *(v8bf*)&As[ar][ak + 8] = *(const v8bf*)(ga + 8);

    const bf16* gb = W + (size_t)(kt + bkr) * Ntot + n0 + bnc;
    v8bf w0 = *(const v8bf*)gb;
    v8bf w1 = *(const v8bf*)(gb + 8);
#pragma unroll
    for (int j = 0; j < 8; ++j) {
      Bs[bnc + j][bkr]     = w0[j];
      Bs[bnc + 8 + j][bkr] = w1[j];
    }

    // prefetch next K tile into caches (global_prefetch_b8)
    if (kt + BK < Ktot) {
      __builtin_prefetch(ga + BK, 0, 1);
      __builtin_prefetch(gb + (size_t)BK * Ntot, 0, 1);
    }
    __syncthreads();

    // --- fragments + WMMA ---
    v16bf afr[2], bfr[4];
#pragma unroll
    for (int i = 0; i < 2; ++i) {
      const bf16* p = &As[wr + i * 16 + lr][khalf];
      afr[i] = cat8(*(const v8bf*)p, *(const v8bf*)(p + 16));
    }
#pragma unroll
    for (int j = 0; j < 4; ++j) {
      const bf16* p = &Bs[wc + j * 16 + lr][khalf];
      bfr[j] = cat8(*(const v8bf*)p, *(const v8bf*)(p + 16));
    }
#pragma unroll
    for (int i = 0; i < 2; ++i)
#pragma unroll
      for (int j = 0; j < 4; ++j) acc[i][j] = wmma_bf16(afr[i], bfr[j], acc[i][j]);

    __syncthreads();
  }

  // --- epilogue: bias, scale, store (C/D layout: lane<16 rows 0..7, lane>=16 rows 8..15)
  const int rowHalf = (lane >> 4) * 8;
#pragma unroll
  for (int i = 0; i < 2; ++i)
#pragma unroll
    for (int j = 0; j < 4; ++j) {
      const int gcol = n0 + wc + j * 16 + lr;
      const float bv = bias[gcol];
#pragma unroll
      for (int r = 0; r < 8; ++r) {
        const int grow = m0 + wr + i * 16 + rowHalf + r;
        const float val = (acc[i][j][r] + bv) * scale;
        if (OUT_F32)
          ((float*)Cout)[(size_t)grow * Ntot + gcol] = val;
        else
          ((bf16*)Cout)[(size_t)grow * Ntot + gcol] = (bf16)val;
      }
    }
}

// ---------------------------------------------------------------------------
// TDM: issue a tensor_load_to_lds of a 64x64 bf16 tile (row stride CC elems)
// into LDS with 16B row padding (64 data + 8 pad bf16 -> 72-elem rows).
// Descriptor per CDNA5 ISA ch.8 (D# groups). Issued by one wave; completion
// via s_wait_tensorcnt + workgroup barrier.
// ---------------------------------------------------------------------------
#if HAVE_TDM
__device__ __forceinline__ void tdm_load_tile64x64(unsigned ldsByteAddr,
                                                   const bf16* gsrc) {
  const unsigned long long gaddr = (unsigned long long)(uintptr_t)gsrc;

  u32x4 g0;
  g0[0] = 1u;                                     // count=1 (valid), user mode
  g0[1] = ldsByteAddr;                            // lds_addr
  g0[2] = (unsigned)(gaddr & 0xffffffffull);      // global_addr[31:0]
  g0[3] = (unsigned)((gaddr >> 32) & 0x01ffffffull) | (2u << 30); // addr[56:32] | type=2

  i32x8 g1;
  // data_size=1 (2B) @16; pad_enable @20; pad_interval=4 (32 dwords) @22;
  // pad_amount=3 (4 dwords = 16B) @25; workgroup_mask=0
  g1[0] = (1 << 16) | (1 << 20) | (4 << 22) | (3 << 25);
  g1[1] = (int)((CC & 0xffff) << 16);             // atomic_barrier=0 | tensor_dim0.lo
  g1[2] = (int)(((CC >> 16) & 0xffff) | ((MROWS & 0xffff) << 16)); // dim0.hi | dim1.lo
  g1[3] = (int)(((MROWS >> 16) & 0xffff) | (64u << 16));           // dim1.hi | tile_dim0=64
  g1[4] = 64;                                     // tile_dim1=64, tile_dim2=0
  g1[5] = CC;                                     // tensor_dim0_stride[31:0]
  g1[6] = 0;                                      // stride.hi | dim1_stride.lo (2D)
  g1[7] = 0;

  i32x4 gz;
  gz[0] = 0; gz[1] = 0; gz[2] = 0; gz[3] = 0;

#if defined(__clang_major__) && (__clang_major__ >= 23)
  i32x8 gz8;
#pragma unroll
  for (int i = 0; i < 8; ++i) gz8[i] = 0;
  __builtin_amdgcn_tensor_load_to_lds(g0, g1, gz, gz, gz8, 0);   // 6-arg toolchain
#else
  __builtin_amdgcn_tensor_load_to_lds(g0, g1, gz, gz, 0);        // 5-arg toolchain
#endif
}
#endif

// ---------------------------------------------------------------------------
// Flash-style causal attention.
// Q,K,V bf16 in [B*T, C] layout (head h occupies cols h*64..h*64+63). Q pre-
// scaled by 1/sqrt(D). Grid: (T/128, NHEAD, B). 256 thr = 8 waves; each wave
// owns 16 queries; key blocks of 64 staged in LDS (K tile via TDM).
// ---------------------------------------------------------------------------
__global__ void __launch_bounds__(256)
attn_causal_kernel(const bf16* __restrict__ Q, const bf16* __restrict__ Kg,
                   const bf16* __restrict__ Vg, bf16* __restrict__ ctx) {
  __shared__ __align__(16) bf16 Kt[64][72];        // [key][d]    (natural)
  __shared__ __align__(16) bf16 Vt[64][72];        // [d][key]    (transposed)
  __shared__ __align__(16) bf16 Pst[8][16][72];    // per-wave P staging [q][s]

  const int tid  = threadIdx.x;
  const int lane = tid & 31;
  const int wave = tid >> 5;
  const int qblk  = blockIdx.x * 128;
  const int head  = blockIdx.y;
  const int batch = blockIdx.z;
  const int qw    = qblk + wave * 16;          // first query row of this wave
  const size_t baseBT = (size_t)batch * TT;

  const int lr      = lane & 15;
  const int khalf   = (lane >> 4) * 8;
  const int ks16    = (lane >> 4) * 16;
  const int rowHalf = (lane >> 4) * 8;

  // Q fragments (A layout), d contracted in two chunks of 32
  v16bf qf[2];
  {
    const bf16* qrow = Q + (baseBT + qw + lr) * CC + head * HDIM;
#pragma unroll
    for (int f = 0; f < 2; ++f) {
      const bf16* p = qrow + f * 32 + khalf;
      qf[f] = cat8(*(const v8bf*)p, *(const v8bf*)(p + 16));
    }
  }

  v8f o[4];
#pragma unroll
  for (int dg = 0; dg < 4; ++dg) o[dg] = v8f_zero();
  float mstate[8], lstate[8];
#pragma unroll
  for (int r = 0; r < 8; ++r) { mstate[r] = -__builtin_inff(); lstate[r] = 0.0f; }

  // tile loader indices: 4 threads per key row, 16 d-elems each
  const int trow = tid >> 2;
  const int tcol = (tid & 3) * 16;

#if HAVE_TDM
  const unsigned ldsK = (unsigned)(uintptr_t)&Kt[0][0];
#endif

  for (int s0 = 0; s0 < qblk + 128; s0 += 64) {
    // ---- stage K tile: TDM async DMA (wave 0 issues, waits on TENSORcnt) ----
#if HAVE_TDM
    if (tid < 32) {
      tdm_load_tile64x64(ldsK, Kg + (baseBT + s0) * CC + head * HDIM);
      __builtin_amdgcn_s_wait_tensorcnt(0);
    }
#else
    {
      const bf16* krow = Kg + (baseBT + s0 + trow) * CC + head * HDIM + tcol;
      *(v8bf*)&Kt[trow][tcol]     = *(const v8bf*)krow;
      *(v8bf*)&Kt[trow][tcol + 8] = *(const v8bf*)(krow + 8);
    }
#endif
    // ---- stage V tile transposed (all threads) ----
    {
      const bf16* vrow = Vg + (baseBT + s0 + trow) * CC + head * HDIM + tcol;
      v8bf v0 = *(const v8bf*)vrow;
      v8bf v1 = *(const v8bf*)(vrow + 8);
#pragma unroll
      for (int j = 0; j < 8; ++j) {
        Vt[tcol + j][trow]     = v0[j];
        Vt[tcol + 8 + j][trow] = v1[j];
      }
    }
    __syncthreads();

    if (s0 <= qw + 15) {   // this wave has at least one unmasked key here
      // ---- scores: 16 queries x 64 keys ----
      v8f sc[4];
#pragma unroll
      for (int kg = 0; kg < 4; ++kg) {
        sc[kg] = v8f_zero();
#pragma unroll
        for (int dc = 0; dc < 2; ++dc) {
          const bf16* p = &Kt[kg * 16 + lr][dc * 32 + khalf];
          v16bf bfrag = cat8(*(const v8bf*)p, *(const v8bf*)(p + 16));
          sc[kg] = wmma_bf16(qf[dc], bfrag, sc[kg]);
        }
      }
      // ---- causal mask (diagonal blocks only) ----
      if (s0 + 63 > qw) {
#pragma unroll
        for (int kg = 0; kg < 4; ++kg) {
          const int ncol = s0 + kg * 16 + lr;
#pragma unroll
          for (int r = 0; r < 8; ++r)
            if (ncol > qw + rowHalf + r) sc[kg][r] = -__builtin_inff();
        }
      }
      // ---- row max (across 16 lanes of each half-wave) ----
      float pmax[8];
#pragma unroll
      for (int r = 0; r < 8; ++r) {
        float v = sc[0][r];
#pragma unroll
        for (int kg = 1; kg < 4; ++kg) v = fmaxf(v, sc[kg][r]);
        pmax[r] = v;
      }
#pragma unroll
      for (int off = 1; off < 16; off <<= 1)
#pragma unroll
        for (int r = 0; r < 8; ++r)
          pmax[r] = fmaxf(pmax[r], __shfl_xor(pmax[r], off, 32));

      // ---- online softmax update ----
      float corr[8], psum[8];
#pragma unroll
      for (int r = 0; r < 8; ++r) {
        const float mnew = fmaxf(mstate[r], pmax[r]);
        corr[r] = __expf(mstate[r] - mnew);
        mstate[r] = mnew;
        psum[r] = 0.0f;
      }
#pragma unroll
      for (int kg = 0; kg < 4; ++kg)
#pragma unroll
        for (int r = 0; r < 8; ++r) {
          const float p = __expf(sc[kg][r] - mstate[r]);
          sc[kg][r] = p;
          psum[r] += p;
        }
#pragma unroll
      for (int off = 1; off < 16; off <<= 1)
#pragma unroll
        for (int r = 0; r < 8; ++r) psum[r] += __shfl_xor(psum[r], off, 32);
#pragma unroll
      for (int r = 0; r < 8; ++r) lstate[r] = lstate[r] * corr[r] + psum[r];
#pragma unroll
      for (int dg = 0; dg < 4; ++dg)
#pragma unroll
        for (int r = 0; r < 8; ++r) o[dg][r] *= corr[r];

      // ---- P: C-layout -> A-layout via per-wave LDS staging ----
#pragma unroll
      for (int kg = 0; kg < 4; ++kg)
#pragma unroll
        for (int r = 0; r < 8; ++r)
          Pst[wave][rowHalf + r][kg * 16 + lr] = (bf16)sc[kg][r];

      // ---- O += P @ V ----
#pragma unroll
      for (int dg = 0; dg < 4; ++dg) {
#pragma unroll
        for (int sch = 0; sch < 2; ++sch) {
          const bf16* pa = &Pst[wave][lr][sch * 32 + khalf];
          v16bf af = cat8(*(const v8bf*)pa, *(const v8bf*)(pa + 16));
          const bf16* pb = &Vt[dg * 16 + lr][sch * 32 + khalf];
          v16bf bf = cat8(*(const v8bf*)pb, *(const v8bf*)(pb + 16));
          o[dg] = wmma_bf16(af, bf, o[dg]);
        }
      }
    }
    __syncthreads();
  }

  // ---- normalize and store context (bf16, [B*T, C] with head column offset)
#pragma unroll
  for (int dg = 0; dg < 4; ++dg) {
    const int d = head * HDIM + dg * 16 + lr;
#pragma unroll
    for (int r = 0; r < 8; ++r) {
      const int m = qw + rowHalf + r;
      ctx[(baseBT + m) * CC + d] = (bf16)(o[dg][r] / lstate[r]);
    }
  }
}

// ---------------------------------------------------------------------------
// Launch: convert -> QKV GEMMs -> causal attention -> output GEMM
// ---------------------------------------------------------------------------
extern "C" void kernel_launch(void* const* d_in, const int* in_sizes, int n_in,
                              void* d_out, int out_size, void* d_ws, size_t ws_size,
                              hipStream_t stream) {
  const float* src = (const float*)d_in[0];
  // d_in[1] = mask (unused on the causal path)
  const float* Wq = (const float*)d_in[2];
  const float* bq = (const float*)d_in[3];
  const float* Wk = (const float*)d_in[4];
  const float* bk = (const float*)d_in[5];
  const float* Wv = (const float*)d_in[6];
  const float* bv = (const float*)d_in[7];
  const float* Wo = (const float*)d_in[8];
  const float* bo = (const float*)d_in[9];
  float* out = (float*)d_out;

  char* ws = (char*)d_ws;
  bf16* sBF  = (bf16*)(ws);                        // 4096x1024  (8 MB)
  bf16* wqBF = (bf16*)(ws + (size_t)( 8u << 20));  // 1024x1024  (2 MB each)
  bf16* wkBF = (bf16*)(ws + (size_t)(10u << 20));
  bf16* wvBF = (bf16*)(ws + (size_t)(12u << 20));
  bf16* woBF = (bf16*)(ws + (size_t)(14u << 20));
  bf16* qBF  = (bf16*)(ws + (size_t)(16u << 20));  // 8 MB each
  bf16* kBF  = (bf16*)(ws + (size_t)(24u << 20));
  bf16* vBF  = (bf16*)(ws + (size_t)(32u << 20));
  bf16* ctx  = (bf16*)(ws + (size_t)(40u << 20));  // total 48 MB

  // fp32 -> bf16
  f32_to_bf16_kernel<<<2048, 256, 0, stream>>>(src, sBF, MROWS * CC);
  f32_to_bf16_kernel<<< 512, 256, 0, stream>>>(Wq,  wqBF, CC * CC);
  f32_to_bf16_kernel<<< 512, 256, 0, stream>>>(Wk,  wkBF, CC * CC);
  f32_to_bf16_kernel<<< 512, 256, 0, stream>>>(Wv,  wvBF, CC * CC);
  f32_to_bf16_kernel<<< 512, 256, 0, stream>>>(Wo,  woBF, CC * CC);

  // QKV projections (Q pre-scaled by 1/sqrt(64) = 0.125)
  dim3 gg(CC / BN, MROWS / BM);   // (8, 32)
  gemm_bf16_kernel<0><<<gg, 256, 0, stream>>>(sBF, wqBF, bq, qBF, MROWS, CC, CC, 0.125f);
  gemm_bf16_kernel<0><<<gg, 256, 0, stream>>>(sBF, wkBF, bk, kBF, MROWS, CC, CC, 1.0f);
  gemm_bf16_kernel<0><<<gg, 256, 0, stream>>>(sBF, wvBF, bv, vBF, MROWS, CC, CC, 1.0f);

  // causal flash attention (K tiles via Tensor Data Mover when available)
  dim3 ga(TT / 128, NHEAD, 2);    // (16, 16, 2)
  attn_causal_kernel<<<ga, 256, 0, stream>>>(qBF, kBF, vBF, ctx);

  // output projection -> fp32 d_out
  gemm_bf16_kernel<1><<<gg, 256, 0, stream>>>(ctx, woBF, bo, out, MROWS, CC, CC, 1.0f);
}
